// LeNet5XNOR_86045374808886
// MI455X (gfx1250) — matrix operand order repaired and checked by reading, not compiled, not run on hardware
//
#include <hip/hip_runtime.h>
#include <hip/hip_bf16.h>

// ---------------------------------------------------------------------------
// LeNet5-XNOR forward, fused for MI455X (gfx1250, wave32, WMMA f16 16x16x32).
// Binary convs / linear are exact in f16 WMMA (+/-1 representable, f32 accum).
// v2: all WMMA operands stored in *fragment-ready* per-lane layout ->
//     one contiguous 32B v16h load per lane per fragment (ds_load_b128 pairs)
//     instead of 16x ds_load_u16 + v_mov_b16 packing.
// ---------------------------------------------------------------------------

typedef __attribute__((ext_vector_type(16))) _Float16 v16h;
typedef __attribute__((ext_vector_type(8)))  float    v8f;

#define BN_EPS 1e-4f
#define BATCH  8192

__device__ __forceinline__ v8f wmma_f16(v16h a, v16h b, v8f c) {
  return __builtin_amdgcn_wmma_f32_16x16x32_f16(false, a, false, b,
                                                (short)0, c, false, false);
}

__device__ __forceinline__ _Float16 bsign(float x) {
  return x >= 0.f ? (_Float16)1.f : (_Float16)-1.f;
}

// ---- K0: zero stats, build weights in WMMA A-fragment-ready layout --------
// dst[((mt*Ksteps + ks)*32 + lane)*16 + e]; lane holds row mt*16+(lane&15),
// element e -> K = ks*32 + 16*(lane>>4) + e.
__device__ __forceinline__ void build_wfrag(_Float16* dst, const float* src,
                                            int Mvalid, int Kvalid,
                                            int Mtiles, int Ksteps, bool binar,
                                            int t, int stride) {
  const int total = Mtiles * Ksteps * 32 * 16;
  for (int idx = t; idx < total; idx += stride) {
    int e    = idx & 15;
    int fr   = idx >> 4;
    int lane = fr & 31;
    int mtks = fr >> 5;
    int ks   = mtks % Ksteps, mt = mtks / Ksteps;
    int row  = mt * 16 + (lane & 15);
    int kk   = ks * 32 + (lane >> 4) * 16 + e;
    _Float16 v = (_Float16)0.f;
    if (row < Mvalid && kk < Kvalid) {
      float wv = src[row * Kvalid + kk];
      v = binar ? bsign(wv) : (_Float16)wv;
    }
    dst[idx] = v;
  }
}

__global__ void k0_prep(const float* c2w, const float* c3w,
                        const float* linw, const float* fcw,
                        float* stats, _Float16* Wc2, _Float16* Wc3,
                        _Float16* Wlin, _Float16* Wfc) {
  const int stride = blockDim.x * gridDim.x;
  const int t = blockIdx.x * blockDim.x + threadIdx.x;
  for (int i = t; i < 1024; i += stride) stats[i] = 0.f;
  build_wfrag(Wc2,  c2w,  16, 150, 1,  5, true,  t, stride);
  build_wfrag(Wc3,  c3w, 120, 400, 8, 13, true,  t, stride);
  build_wfrag(Wlin, linw, 84, 120, 6,  4, true,  t, stride);
  build_wfrag(Wfc,  fcw,  10,  84, 1,  3, false, t, stride);
}

// ---- K1: conv1, accumulate per-channel sum/sumsq --------------------------
__global__ void k1_conv1_stats(const float* __restrict__ x,
                               const float* __restrict__ w,
                               const float* __restrict__ bias,
                               float* sum, float* sq) {
  __shared__ float ls[6], lq[6];
  if (threadIdx.x < 6) { ls[threadIdx.x] = 0.f; lq[threadIdx.x] = 0.f; }
  __syncthreads();
  const long long gid = (long long)blockIdx.x * blockDim.x + threadIdx.x;
  const long long NP  = (long long)BATCH * 28 * 28;
  if (gid < NP) {
    int b = (int)(gid / 784), p = (int)(gid % 784);
    int oy = p / 28, ox = p % 28;
    const float* xp = x + (long long)b * 1024 + oy * 32 + ox;
    __builtin_prefetch(xp + 32, 0, 0);
    float patch[25];
#pragma unroll
    for (int ky = 0; ky < 5; ++ky)
#pragma unroll
      for (int kx = 0; kx < 5; ++kx) patch[ky * 5 + kx] = xp[ky * 32 + kx];
#pragma unroll
    for (int c = 0; c < 6; ++c) {
      float a = bias[c];
#pragma unroll
      for (int k = 0; k < 25; ++k) a = fmaf(patch[k], w[c * 25 + k], a);
      atomicAdd(&ls[c], a);
      atomicAdd(&lq[c], a * a);
    }
  }
  __syncthreads();
  if (threadIdx.x < 6) {
    atomicAdd(&sum[threadIdx.x], ls[threadIdx.x]);
    atomicAdd(&sq[threadIdx.x],  lq[threadIdx.x]);
  }
}

// ---- BN finalize: sum/sq -> (scale, shift) in place -----------------------
__global__ void k_bnfin(float* sum, float* sq, const float* w, const float* b,
                        int C, float invN, float clampw) {
  int c = threadIdx.x;
  if (c < C) {
    float m = sum[c] * invN;
    float v = sq[c] * invN - m * m;
    float r = rsqrtf(v + BN_EPS);
    float ww = 1.f, bb = 0.f;
    if (w) {
      ww = w[c];
      if (clampw > 0.f && ww < clampw) ww = clampw;
      bb = b[c];
    }
    float scale = ww * r;
    sum[c] = scale;           // scale
    sq[c]  = bb - m * scale;  // shift
  }
}

// ---- K3: conv1 + BN1 + ReLU + maxpool2 -> h1, stats for BN2 ---------------
__global__ void k3_layer1(const float* __restrict__ x,
                          const float* __restrict__ w,
                          const float* __restrict__ bias,
                          const float* __restrict__ scale,
                          const float* __restrict__ shift,
                          float* __restrict__ h1, float* sum, float* sq) {
  __shared__ float ls[6], lq[6];
  if (threadIdx.x < 6) { ls[threadIdx.x] = 0.f; lq[threadIdx.x] = 0.f; }
  __syncthreads();
  const long long gid = (long long)blockIdx.x * blockDim.x + threadIdx.x;
  const long long NP  = (long long)BATCH * 14 * 14;
  if (gid < NP) {
    int b = (int)(gid / 196), p = (int)(gid % 196);
    int py = p / 14, px = p % 14;
    const float* xp = x + (long long)b * 1024 + (py * 2) * 32 + px * 2;
    float patch[36];
#pragma unroll
    for (int j = 0; j < 6; ++j)
#pragma unroll
      for (int i = 0; i < 6; ++i) patch[j * 6 + i] = xp[j * 32 + i];
#pragma unroll
    for (int c = 0; c < 6; ++c) {
      float mx = -3.4e38f;
#pragma unroll
      for (int dy = 0; dy < 2; ++dy)
#pragma unroll
        for (int dx = 0; dx < 2; ++dx) {
          float a = bias[c];
#pragma unroll
          for (int ky = 0; ky < 5; ++ky)
#pragma unroll
            for (int kx = 0; kx < 5; ++kx)
              a = fmaf(patch[(dy + ky) * 6 + dx + kx], w[c * 25 + ky * 5 + kx], a);
          float y = fmaxf(a * scale[c] + shift[c], 0.f);  // BN + ReLU
          mx = fmaxf(mx, y);
        }
      h1[((long long)b * 6 + c) * 196 + p] = mx;
      atomicAdd(&ls[c], mx);
      atomicAdd(&lq[c], mx * mx);
    }
  }
  __syncthreads();
  if (threadIdx.x < 6) {
    atomicAdd(&sum[threadIdx.x], ls[threadIdx.x]);
    atomicAdd(&sq[threadIdx.x],  lq[threadIdx.x]);
  }
}

// ---- K5: binary conv2 via WMMA. M=16 oc, K=160 (5 steps), N=B*100 ---------
__global__ void k5_conv2(const float* __restrict__ h1,
                         const _Float16* __restrict__ Wc2,
                         const float* __restrict__ c2b,
                         const float* __restrict__ scale,
                         const float* __restrict__ shift,
                         float* __restrict__ c2out) {
  __shared__ v16h sW[5 * 32];          // weight fragments (shared by 4 waves)
  __shared__ v16h sB[4][5 * 32];       // per-wave activation fragments
  const int tid = threadIdx.x;
  {
    const v16h* Wv = (const v16h*)Wc2;
    for (int i = tid; i < 5 * 32; i += blockDim.x) sW[i] = Wv[i];
  }
  const int wave = tid >> 5, lane = tid & 31;
  const long long NPIX = (long long)BATCH * 100;
  const long long tile = (long long)blockIdx.x * 4 + wave;
  // this lane owns col n and K-half (lane>>4) of every k-step
  const int n  = lane & 15;
  const int kh = (lane >> 4) << 4;     // 0 or 16
  const long long P = tile * 16 + n;
  const bool pin = (P < NPIX);
  const int b  = pin ? (int)(P / 100) : 0;
  const int pp = pin ? (int)(P % 100) : 0;
  const int oy = pp / 10, ox = pp % 10;
#pragma unroll
  for (int ks = 0; ks < 5; ++ks) {
    v16h v;
#pragma unroll
    for (int i = 0; i < 16; ++i) {
      int k = ks * 32 + kh + i;
      _Float16 e = (_Float16)0.f;
      if (k < 150 && pin) {
        int ic = k / 25, r = k % 25, ky = r / 5, kx = r % 5;
        float hv = h1[((long long)b * 6 + ic) * 196 + (oy + ky) * 14 + (ox + kx)];
        e = bsign(hv * scale[ic] + shift[ic]);
      }
      v[i] = e;
    }
    sB[wave][ks * 32 + lane] = v;      // one 32B LDS store
  }
  __syncthreads();
  v8f acc = {};
#pragma unroll
  for (int ks = 0; ks < 5; ++ks)
    acc = wmma_f16(sW[ks * 32 + lane], sB[wave][ks * 32 + lane], acc);
  const int mb = (lane >> 4) * 8;
  if (pin) {
#pragma unroll
    for (int r = 0; r < 8; ++r) {
      int oc = mb + r;
      c2out[((long long)b * 16 + oc) * 100 + pp] = acc[r] + c2b[oc];
    }
  }
}

// ---- K5b: maxpool2 on conv2 output -> h2, stats for BN3 -------------------
__global__ void k5b_pool2(const float* __restrict__ c2out,
                          float* __restrict__ h2, float* sum, float* sq) {
  __shared__ float ls[16], lq[16];
  if (threadIdx.x < 16) { ls[threadIdx.x] = 0.f; lq[threadIdx.x] = 0.f; }
  __syncthreads();
  const long long gid = (long long)blockIdx.x * blockDim.x + threadIdx.x;
  const long long N   = (long long)BATCH * 16 * 25;
  if (gid < N) {
    int b = (int)(gid / 400), rem = (int)(gid % 400);
    int c = rem / 25, p = rem % 25;
    int py = p / 5, px = p % 5;
    const float* src = c2out + ((long long)b * 16 + c) * 100 + py * 20 + px * 2;
    float mx = fmaxf(fmaxf(src[0], src[1]), fmaxf(src[10], src[11]));
    h2[gid] = mx;
    atomicAdd(&ls[c], mx);
    atomicAdd(&lq[c], mx * mx);
  }
  __syncthreads();
  if (threadIdx.x < 16) {
    atomicAdd(&sum[threadIdx.x], ls[threadIdx.x]);
    atomicAdd(&sq[threadIdx.x],  lq[threadIdx.x]);
  }
}

// ---- K7: binary conv3 via WMMA. M=128(120 oc), K=416 (13 steps), N=B ------
__global__ void k7_conv3(const float* __restrict__ h2,
                         const _Float16* __restrict__ Wc3,
                         const float* __restrict__ c3b,
                         const float* __restrict__ scale,
                         const float* __restrict__ shift,
                         float* __restrict__ h3, float* sum, float* sq) {
  __shared__ v16h sB[13 * 32];         // activation fragments, shared by 8 waves
  __shared__ float ls[128], lq[128];
  const int tid = threadIdx.x;         // 256 = 8 waves
  for (int i = tid; i < 128; i += blockDim.x) { ls[i] = 0.f; lq[i] = 0.f; }
  const long long bbase = (long long)blockIdx.x * 16;
  // build B fragments: 13*32 fragment-rows over 256 threads
  for (int fr = tid; fr < 13 * 32; fr += blockDim.x) {
    int ks = fr >> 5, L = fr & 31;
    int n  = L & 15;
    int Kb = ks * 32 + ((L >> 4) << 4);
    const float* hp = h2 + (bbase + n) * 400;   // h2 flat == im2col k order
    v16h v;
#pragma unroll
    for (int i = 0; i < 16; ++i) {
      int k = Kb + i;
      _Float16 e = (_Float16)0.f;
      if (k < 400) {
        int ic = k / 25;
        e = bsign(hp[k] * scale[ic] + shift[ic]);
      }
      v[i] = e;
    }
    sB[fr] = v;
  }
  __syncthreads();
  const int wave = tid >> 5, lane = tid & 31;
  const v16h* Wv = (const v16h*)Wc3;   // L2-resident fragment-ready weights
  v8f acc = {};
#pragma unroll
  for (int ks = 0; ks < 13; ++ks)
    acc = wmma_f16(Wv[(wave * 13 + ks) * 32 + lane], sB[ks * 32 + lane], acc);
  const int n = lane & 15, mb = (lane >> 4) * 8;
#pragma unroll
  for (int r = 0; r < 8; ++r) {
    int oc = wave * 16 + mb + r;
    if (oc < 120) {
      float y = acc[r] + c3b[oc];
      h3[(bbase + n) * 120 + oc] = y;
      atomicAdd(&ls[oc], y);
      atomicAdd(&lq[oc], y * y);
    }
  }
  __syncthreads();
  for (int i = tid; i < 120; i += blockDim.x) {
    atomicAdd(&sum[i], ls[i]);
    atomicAdd(&sq[i],  lq[i]);
  }
}

// ---- K9: BN4 + binarize + WMMA linear(120->84) + WMMA fc(84->10) ----------
__global__ void k9_head(const float* __restrict__ h3,
                        const _Float16* __restrict__ Wlin,
                        const float* __restrict__ linb,
                        const _Float16* __restrict__ Wfc,
                        const float* __restrict__ fcb,
                        const float* __restrict__ scale,
                        const float* __restrict__ shift,
                        float* __restrict__ out) {
  __shared__ v16h sA2[4 * 32];         // binarized activations, frag-ready
  __shared__ v16h sH4[3 * 32];         // linear output, frag-ready
  const int tid = threadIdx.x;         // 192 = 6 waves
  const long long bbase = (long long)blockIdx.x * 16;
  for (int fr = tid; fr < 4 * 32; fr += blockDim.x) {
    int ks = fr >> 5, L = fr & 31;
    int n  = L & 15;
    int Kb = ks * 32 + ((L >> 4) << 4);
    const float* hp = h3 + (bbase + n) * 120;
    v16h v;
#pragma unroll
    for (int i = 0; i < 16; ++i) {
      int k = Kb + i;
      v[i] = (k < 120) ? bsign(hp[k] * scale[k] + shift[k]) : (_Float16)0.f;
    }
    sA2[fr] = v;
  }
  for (int fr = tid; fr < 3 * 32; fr += blockDim.x) {
    v16h z = {};
    sH4[fr] = z;
  }
  __syncthreads();
  const int wave = tid >> 5, lane = tid & 31;
  const int n = lane & 15, mb = (lane >> 4) * 8;
  {  // GEMM1: h4 = Wlin x a2^T  (M=96, K=128, N=16)
    const v16h* Wv = (const v16h*)Wlin;
    v8f acc = {};
#pragma unroll
    for (int ks = 0; ks < 4; ++ks)
      acc = wmma_f16(Wv[(wave * 4 + ks) * 32 + lane], sA2[ks * 32 + lane], acc);
    _Float16* h4h = (_Float16*)sH4;
#pragma unroll
    for (int r = 0; r < 8; ++r) {
      int o = wave * 16 + mb + r;
      if (o < 84) {
        // scatter into B-fragment-ready layout: k=o, col=n
        int fi = (((o >> 5) * 32 + (n + (o & 16))) << 4) + (o & 15);
        h4h[fi] = (_Float16)(acc[r] + linb[o]);
      }
    }
  }
  __syncthreads();
  if (wave == 0) {  // GEMM2: out = Wfc x h4  (M=16, K=96, N=16)
    const v16h* Wv = (const v16h*)Wfc;
    v8f acc = {};
#pragma unroll
    for (int ks = 0; ks < 3; ++ks)
      acc = wmma_f16(Wv[ks * 32 + lane], sH4[ks * 32 + lane], acc);
#pragma unroll
    for (int r = 0; r < 8; ++r) {
      int o = mb + r;
      if (o < 10) out[(bbase + n) * 10 + o] = acc[r] + fcb[o];
    }
  }
}

// ---------------------------------------------------------------------------
extern "C" void kernel_launch(void* const* d_in, const int* in_sizes, int n_in,
                              void* d_out, int out_size, void* d_ws, size_t ws_size,
                              hipStream_t stream) {
  const float* x       = (const float*)d_in[0];
  const float* conv1_w = (const float*)d_in[1];
  const float* conv1_b = (const float*)d_in[2];
  const float* bn2_w   = (const float*)d_in[3];
  const float* bn2_b   = (const float*)d_in[4];
  const float* conv2_w = (const float*)d_in[5];
  const float* conv2_b = (const float*)d_in[6];
  const float* bn3_w   = (const float*)d_in[7];
  const float* bn3_b   = (const float*)d_in[8];
  const float* conv3_w = (const float*)d_in[9];
  const float* conv3_b = (const float*)d_in[10];
  const float* bn4_w   = (const float*)d_in[11];
  const float* bn4_b   = (const float*)d_in[12];
  const float* lin_w   = (const float*)d_in[13];
  const float* lin_b   = (const float*)d_in[14];
  const float* fc_w    = (const float*)d_in[15];
  const float* fc_b    = (const float*)d_in[16];
  float* out = (float*)d_out;

  // workspace carve-up (32B-aligned for v16h access)
  char* ws = (char*)d_ws;
  size_t off = 0;
  float* stats = (float*)(ws + off); off += 1024 * sizeof(float);   // 4 stages x 256
  _Float16* Wc2  = (_Float16*)(ws + off); off += (size_t)1 * 5  * 512 * 2;
  _Float16* Wc3  = (_Float16*)(ws + off); off += (size_t)8 * 13 * 512 * 2;
  _Float16* Wlin = (_Float16*)(ws + off); off += (size_t)6 * 4  * 512 * 2;
  _Float16* Wfc  = (_Float16*)(ws + off); off += (size_t)1 * 3  * 512 * 2;
  off = (off + 255) & ~(size_t)255;
  float* h1    = (float*)(ws + off); off += (size_t)BATCH * 6 * 196 * 4;
  float* c2out = (float*)(ws + off); off += (size_t)BATCH * 16 * 100 * 4;
  float* h2    = (float*)(ws + off); off += (size_t)BATCH * 16 * 25 * 4;
  float* h3    = (float*)(ws + off); off += (size_t)BATCH * 120 * 4;

  float* sum0 = stats + 0 * 256; float* sq0 = sum0 + 128;
  float* sum1 = stats + 1 * 256; float* sq1 = sum1 + 128;
  float* sum2 = stats + 2 * 256; float* sq2 = sum2 + 128;
  float* sum3 = stats + 3 * 256; float* sq3 = sum3 + 128;

  k0_prep<<<64, 256, 0, stream>>>(conv2_w, conv3_w, lin_w, fc_w,
                                  stats, Wc2, Wc3, Wlin, Wfc);

  k1_conv1_stats<<<25088, 256, 0, stream>>>(x, conv1_w, conv1_b, sum0, sq0);
  k_bnfin<<<1, 128, 0, stream>>>(sum0, sq0, nullptr, nullptr, 6,
                                 1.f / (float)((long long)BATCH * 784), 0.f);

  k3_layer1<<<6272, 256, 0, stream>>>(x, conv1_w, conv1_b, sum0, sq0,
                                      h1, sum1, sq1);
  k_bnfin<<<1, 128, 0, stream>>>(sum1, sq1, bn2_w, bn2_b, 6,
                                 1.f / (float)((long long)BATCH * 196), 0.01f);

  k5_conv2<<<12800, 128, 0, stream>>>(h1, Wc2, conv2_b, sum1, sq1, c2out);
  k5b_pool2<<<12800, 256, 0, stream>>>(c2out, h2, sum2, sq2);
  k_bnfin<<<1, 128, 0, stream>>>(sum2, sq2, bn3_w, bn3_b, 16,
                                 1.f / (float)((long long)BATCH * 25), 0.01f);

  k7_conv3<<<512, 256, 0, stream>>>(h2, Wc3, conv3_b, sum2, sq2, h3, sum3, sq3);
  k_bnfin<<<1, 128, 0, stream>>>(sum3, sq3, bn4_w, bn4_b, 120,
                                 1.f / (float)BATCH, 0.f);

  k9_head<<<512, 192, 0, stream>>>(h3, Wlin, lin_b, Wfc, fc_b, sum3, sq3, out);
}
